// WindowSelfAttention_40054865002679
// MI455X (gfx1250) — matrix operand order repaired and checked by reading, not compiled, not run on hardware
//
#include <hip/hip_runtime.h>

typedef _Float16 v16h __attribute__((ext_vector_type(16)));
typedef _Float16 v8h  __attribute__((ext_vector_type(8)));
typedef float    v8f  __attribute__((ext_vector_type(8)));

#define EDIM   384
#define E3DIM  1152
#define TOK    16
#define NSEQ   4096
#define GSIDE  64          // H = W = 64
#define QK_STR 392         // 784B rows: 16B aligned, bank step 4 -> conflict free
#define VT_STR 24          // 48B rows: 16B aligned, bank step 12 -> conflict free
#define AS_STR 24
#define GW     2           // windows per block (dual-accumulator reuse)

// assemble a 16xf16 fragment from two aligned 8xf16 loads
__device__ __forceinline__ v16h ld16(const _Float16* p0, const _Float16* p1) {
    v8h a = *(const v8h*)p0;
    v8h b = *(const v8h*)p1;
    return __builtin_shufflevector(a, b, 0,1,2,3,4,5,6,7,8,9,10,11,12,13,14,15);
}
// low 8 halves from memory, high 8 halves zero (for K-padded fragments)
__device__ __forceinline__ v16h ld8z(const _Float16* p0) {
    v8h a = *(const v8h*)p0;
    v8h z = {};
    return __builtin_shufflevector(a, z, 0,1,2,3,4,5,6,7,8,9,10,11,12,13,14,15);
}
__device__ __forceinline__ v16h zero16() {
    v8h z = {};
    return __builtin_shufflevector(z, z, 0,1,2,3,4,5,6,7,8,9,10,11,12,13,14,15);
}
__device__ __forceinline__ v8f wmma16(v16h a, v16h b, v8f c) {
    return __builtin_amdgcn_wmma_f32_16x16x32_f16(false, a, false, b, (short)0, c,
                                                  false, false);
}
// warm one 768B weight row (16 f16 cols x 384) into the caches
__device__ __forceinline__ void pf_row(const _Float16* p) {
#pragma unroll
    for (int pf = 0; pf < 6; ++pf)
        __builtin_prefetch((const void*)(p + pf * 64), 0, 3);   // 128B steps
}

__global__ void wsa_cvt_f16(const float* __restrict__ src, _Float16* __restrict__ dst,
                            int n) {
    int i = blockIdx.x * 256 + threadIdx.x;
    if (i < n) dst[i] = (_Float16)src[i];
}

// One block = two 4x4 windows. 4 waves of 32. Each wave holds A fragments for
// BOTH windows in VGPRs and issues two WMMAs per weight B-fragment load.
// Next-tile weight rows are prefetched so the B-stream hits near cache.
__global__ __launch_bounds__(128) void wsa_fused(
    const float* __restrict__ x,            // [B,4096,384] f32
    const _Float16* __restrict__ w_in_h,    // [1152,384] f16
    const float* __restrict__ b_in,         // [1152] f32
    const _Float16* __restrict__ w_out_h,   // [384,384] f16
    const float* __restrict__ b_out,        // [384] f32
    float* __restrict__ out)                // [B,4096,384] f32
{
    // pool: x staging [GW][16][QK_STR] (phase 1-2a), then vT [GW][384][VT_STR]
    __shared__ __align__(16) _Float16 pool[GW * EDIM * VT_STR];   // 36864 B
    // qS doubles as attention-output buffer (per-head dims are consumed by the
    // owning wave before being overwritten)
    __shared__ __align__(16) _Float16 qS[GW][TOK][QK_STR];        // 25088 B
    __shared__ __align__(16) _Float16 kS[GW][TOK][QK_STR];        // 25088 B
    __shared__ __align__(16) _Float16 aS[4][TOK][AS_STR];         //  3072 B

    const int tid  = threadIdx.x;
    const int wave = tid >> 5;
    const int lane = tid & 31;
    const int m    = lane & 15;
    const int hi   = lane >> 4;
    const int wid0 = blockIdx.x * GW;

    // warm the first QKV weight tile for this wave while x is being staged
    pf_row(w_in_h + (size_t)(wave * 18 * 16 + m) * EDIM);

    // ---- Phase 1: stage x tokens (f16) for both windows -------------------
    for (int i = tid; i < GW * TOK * EDIM; i += 128) {
        int g  = i / (TOK * EDIM);
        int rr = i - g * (TOK * EDIM);
        int t  = rr / EDIM, d = rr - t * EDIM;
        int wid = wid0 + g;
        int bb = wid >> 8, rem = wid & 255;
        int wh = rem >> 4, ww = rem & 15;
        int r = t >> 2, c = t & 3;
        int nidx = (wh * 4 + r) * GSIDE + (ww * 4 + c);
        pool[g * (TOK * QK_STR) + t * QK_STR + d] =
            (_Float16)x[((size_t)bb * NSEQ + nidx) * EDIM + d];
    }
    __syncthreads();

    // ---- Phase 2a: A fragments for both windows -> VGPRs ------------------
    v16h af[GW][12];
#pragma unroll
    for (int g = 0; g < GW; ++g)
#pragma unroll
        for (int kt = 0; kt < 12; ++kt) {
            const _Float16* row = &pool[g * (TOK * QK_STR) + m * QK_STR + kt * 32];
            af[g][kt] = ld16(row + hi * 8, row + 16 + hi * 8);
        }
    __syncthreads();   // pool is about to be overwritten by vT

    // ---- Phase 2b: QKV projection, two windows per B-fragment load --------
    // 72 N-tiles of 16 across 4 waves. Tiles 0-23:q 24-47:k 48-71:v
    for (int i = 0; i < 18; ++i) {
        int nt   = wave * 18 + i;
        int ncol = nt * 16 + m;
        const _Float16* wr = w_in_h + (size_t)ncol * EDIM + hi * 8;
        float bias = b_in[ncol];
        if (i + 1 < 18)                            // warm next tile's row
            pf_row(w_in_h + (size_t)(ncol + 16) * EDIM);
        v8f c0 = {}, c1 = {};
#pragma unroll
        for (int kt = 0; kt < 12; ++kt) {
            v16h b = ld16(wr + kt * 32, wr + kt * 32 + 16);
            c0 = wmma16(af[0][kt], b, c0);
            c1 = wmma16(af[1][kt], b, c1);
        }
#pragma unroll
        for (int g = 0; g < GW; ++g) {
            v8f c = g ? c1 : c0;
            if (nt < 24) {
                int dim = ncol;
#pragma unroll
                for (int r = 0; r < 8; ++r)
                    qS[g][r + hi * 8][dim] = (_Float16)(c[r] + bias);
            } else if (nt < 48) {
                int dim = ncol - EDIM;
#pragma unroll
                for (int r = 0; r < 8; ++r)
                    kS[g][r + hi * 8][dim] = (_Float16)(c[r] + bias);
            } else {
                int dim = ncol - 2 * EDIM;          // vT[g][dim][token]
#pragma unroll
                for (int r = 0; r < 8; ++r)
                    pool[g * (EDIM * VT_STR) + dim * VT_STR + r + hi * 8] =
                        (_Float16)(c[r] + bias);
            }
        }
    }
    __syncthreads();

    // warm this wave's w_out rows during the attention phase
#pragma unroll
    for (int i = 0; i < 6; ++i)
        pf_row(w_out_h + (size_t)((wave * 6 + i) * 16 + m) * EDIM);

    // ---- Phase 3: attention, 3 heads per wave, both windows ---------------
    const float scale = 0.17677669529663687f;          // 1/sqrt(32)
    for (int hh = 0; hh < 3; ++hh) {
        int h  = wave * 3 + hh;
        int hb = h * 32;
#pragma unroll
        for (int g = 0; g < GW; ++g) {
            // scores[16,16] = (Q slice 16x32) @ (K slice)^T : one wmma
            v16h qa = ld16(&qS[g][m][hb + hi * 8], &qS[g][m][hb + 16 + hi * 8]);
            v16h kb = ld16(&kS[g][m][hb + hi * 8], &kS[g][m][hb + 16 + hi * 8]);
            v8f s = {};
            s = wmma16(qa, kb, s);

            // softmax along N (16 lanes of each half-group hold one row)
#pragma unroll
            for (int r = 0; r < 8; ++r) {
                float v = s[r] * scale;
                float mx = v;
                mx = fmaxf(mx, __shfl_xor(mx, 1));
                mx = fmaxf(mx, __shfl_xor(mx, 2));
                mx = fmaxf(mx, __shfl_xor(mx, 4));
                mx = fmaxf(mx, __shfl_xor(mx, 8));
                float e = __expf(v - mx);
                float sum = e;
                sum += __shfl_xor(sum, 1);
                sum += __shfl_xor(sum, 2);
                sum += __shfl_xor(sum, 4);
                sum += __shfl_xor(sum, 8);
                s[r] = e / sum;
            }

            // re-layout attn (D layout) -> A fragment via per-wave LDS tile
#pragma unroll
            for (int r = 0; r < 8; ++r)
                aS[wave][r + hi * 8][m] = (_Float16)s[r];
            __threadfence_block();                  // order intra-wave LDS st->ld

            // A = attn padded 16x32 (K>=16 zero); hi-half B fragments are zero
            v16h pa = hi ? zero16() : ld16(&aS[wave][m][0], &aS[wave][m][8]);
#pragma unroll
            for (int j = 0; j < 2; ++j) {
                v16h vb = zero16();
                if (hi == 0) {
                    const _Float16* vr =
                        &pool[g * (EDIM * VT_STR) + (hb + j * 16 + m) * VT_STR];
                    vb = ld16(vr, vr + 8);
                }
                v8f o = {};
                o = wmma16(pa, vb, o);
                int dim = hb + j * 16 + m;
#pragma unroll
                for (int r = 0; r < 8; ++r)
                    qS[g][r + hi * 8][dim] = (_Float16)o[r];  // o aliases q
            }
        }
    }
    __syncthreads();

    // ---- Phase 4: out projection, two windows per B-fragment load ---------
    v16h oa[GW][12];
#pragma unroll
    for (int g = 0; g < GW; ++g)
#pragma unroll
        for (int kt = 0; kt < 12; ++kt) {
            const _Float16* row = &qS[g][m][kt * 32];
            oa[g][kt] = ld16(row + hi * 8, row + 16 + hi * 8);
        }
    for (int i = 0; i < 6; ++i) {
        int nt   = wave * 6 + i;
        int ncol = nt * 16 + m;
        const _Float16* wr = w_out_h + (size_t)ncol * EDIM + hi * 8;
        float bias = b_out[ncol];
        if (i + 1 < 6)                             // warm next tile's row
            pf_row(w_out_h + (size_t)(ncol + 16) * EDIM);
        v8f c0 = {}, c1 = {};
#pragma unroll
        for (int kt = 0; kt < 12; ++kt) {
            v16h b = ld16(wr + kt * 32, wr + kt * 32 + 16);
            c0 = wmma16(oa[0][kt], b, c0);
            c1 = wmma16(oa[1][kt], b, c1);
        }
#pragma unroll
        for (int g = 0; g < GW; ++g) {
            v8f c = g ? c1 : c0;
            int wid = wid0 + g;
            int bb = wid >> 8, rem = wid & 255;
            int wh = rem >> 4, ww = rem & 15;
#pragma unroll
            for (int r = 0; r < 8; ++r) {
                int t = r + hi * 8;
                int rr = t >> 2, cc = t & 3;
                int nidx = (wh * 4 + rr) * GSIDE + (ww * 4 + cc);
                out[((size_t)bb * NSEQ + nidx) * EDIM + ncol] = c[r] + bias;
            }
        }
    }
}

extern "C" void kernel_launch(void* const* d_in, const int* in_sizes, int n_in,
                              void* d_out, int out_size, void* d_ws, size_t ws_size,
                              hipStream_t stream) {
    const float* x     = (const float*)d_in[0];
    const float* w_in  = (const float*)d_in[1];
    const float* b_in  = (const float*)d_in[2];
    const float* w_out = (const float*)d_in[3];
    const float* b_out = (const float*)d_in[4];

    const int n_win_w = E3DIM * EDIM;   // 442368
    const int n_wout  = EDIM * EDIM;    // 147456
    _Float16* w_in_h  = (_Float16*)d_ws;
    _Float16* w_out_h = w_in_h + n_win_w;

    wsa_cvt_f16<<<(n_win_w + 255) / 256, 256, 0, stream>>>(w_in, w_in_h, n_win_w);
    wsa_cvt_f16<<<(n_wout + 255) / 256, 256, 0, stream>>>(w_out, w_out_h, n_wout);

    int B = in_sizes[0] / (NSEQ * EDIM);          // 32
    int nblocks = (B * 256) / GW;                 // 4096 blocks, 2 windows each
    wsa_fused<<<nblocks, 128, 0, stream>>>(x, w_in_h, b_in, w_out_h, b_out,
                                           (float*)d_out);
}
